// MaskModel_79714593014104
// MI455X (gfx1250) — matrix operand order repaired and checked by reading, not compiled, not run on hardware
//
#include <hip/hip_runtime.h>
#include <hip/hip_bf16.h>

// ---------------------------------------------------------------------------
// CDNA5 (gfx1250) wave32 WMMA implementation of the mask head.
// Fragment layouts per CDNA5 ISA 7.12.2:
//   A 16x32 f16 : lane l -> row M=l&15 ; halves = K runs [kg*8,+8) and [16+kg*8,+8), kg=l>>4
//   B 32x16 f16 : lane l -> col N=l&15 ; halves = K runs [kg*8,+8) and [16+kg*8,+8)   (weights stored [N][K])
//   C/D 16x16 f32: lane l -> col N=l&15 ; vgpr v -> row M = v + 8*(l>>4)
// ---------------------------------------------------------------------------

typedef __attribute__((ext_vector_type(16))) _Float16 v16h;
typedef __attribute__((ext_vector_type(8)))  float    v8f;
typedef __attribute__((ext_vector_type(4)))  float    f4;
typedef __attribute__((ext_vector_type(4)))  _Float16 h4;

union Frag {
  v16h v;
  f4   q[2];
};

#define NBOX 4096
#define HW   512

// ---------------------------------------------------------------------------
// Weight convert + transpose: w [K][N] f32 -> wT [N][K] f16
// ---------------------------------------------------------------------------
__global__ void convert_transpose(const float* __restrict__ w,
                                  _Float16* __restrict__ wT,
                                  int K, int N) {
  int idx = blockIdx.x * blockDim.x + threadIdx.x;
  int total = K * N;
  if (idx >= total) return;
  int k = idx / N;
  int n = idx - k * N;
  wT[(size_t)n * K + k] = (_Float16)w[idx];
}

// ---------------------------------------------------------------------------
// Fused: crop_and_resize(features)*crop_and_resize(proposal) -> LDS (16x16x64)
//        conv1 3x3 s2 SAME 64->128 relu  -> LDS (8x8x128)
//        conv2 3x3 s2 SAME 128->128 relu -> global y2 [box][16][128] (=[box][2048])
// One workgroup (256 threads / 8 waves) per box.
// ---------------------------------------------------------------------------
__global__ void __launch_bounds__(256)
fused_crop_conv(const float* __restrict__ features,
                const float* __restrict__ proposal,
                const float* __restrict__ bboxes,
                const int*   __restrict__ box_idx,
                const _Float16* __restrict__ w1T,   // [128][576]
                const float* __restrict__ b1,
                const _Float16* __restrict__ w2T,   // [128][1152]
                const float* __restrict__ b2,
                _Float16* __restrict__ y2out)       // [NBOX][2048]
{
  __shared__ __align__(16) _Float16 smx[16 * 16 * 64];  // 32 KB crop tile
  __shared__ __align__(16) _Float16 smy[8 * 8 * 128];   // 16 KB conv1 out

  const int box = blockIdx.x;
  const int t   = threadIdx.x;

  // ---------------- stage 1: bilinear crop * proposal crop -> smx ----------
  {
    const int py = t >> 4, px = t & 15;
    const float by1 = bboxes[box * 4 + 0];
    const float bx1 = bboxes[box * 4 + 1];
    const float by2 = bboxes[box * 4 + 2];
    const float bx2 = bboxes[box * 4 + 3];
    const int   b   = box_idx[box];

    const float ys = (by1 + (float)py * (1.0f / 15.0f) * (by2 - by1)) * 511.0f;
    const float xs = (bx1 + (float)px * (1.0f / 15.0f) * (bx2 - bx1)) * 511.0f;
    const float y0f = floorf(ys), x0f = floorf(xs);
    const float wy = ys - y0f, wx = xs - x0f;
    const int y0  = min(max((int)y0f, 0), HW - 1);
    const int y1i = min(y0 + 1, HW - 1);
    const int x0  = min(max((int)x0f, 0), HW - 1);
    const int x1i = min(x0 + 1, HW - 1);

    const size_t rb = (size_t)b * HW * HW;
    const size_t i00 = rb + (size_t)y0  * HW + x0;
    const size_t i01 = rb + (size_t)y0  * HW + x1i;
    const size_t i10 = rb + (size_t)y1i * HW + x0;
    const size_t i11 = rb + (size_t)y1i * HW + x1i;

    // proposal (single channel) bilinear
    const float p = (proposal[i00] * (1.f - wx) + proposal[i01] * wx) * (1.f - wy) +
                    (proposal[i10] * (1.f - wx) + proposal[i11] * wx) * wy;

    const float w00 = (1.f - wy) * (1.f - wx) * p;
    const float w01 = (1.f - wy) * wx * p;
    const float w10 = wy * (1.f - wx) * p;
    const float w11 = wy * wx * p;

    const float* f00 = features + (i00 << 6);
    const float* f01 = features + (i01 << 6);
    const float* f10 = features + (i10 << 6);
    const float* f11 = features + (i11 << 6);

    #pragma unroll
    for (int c = 0; c < 64; c += 4) {
      f4 a = *(const f4*)(f00 + c);
      f4 bq = *(const f4*)(f01 + c);
      f4 cq = *(const f4*)(f10 + c);
      f4 dq = *(const f4*)(f11 + c);
      f4 r = a * w00 + bq * w01 + cq * w10 + dq * w11;
      *(h4*)&smx[t * 64 + c] = __builtin_convertvector(r, h4);
    }
  }
  __syncthreads();

  const int lane = t & 31;
  const int wv   = t >> 5;       // wave id = N-tile (0..7)
  const int lr   = lane & 15;
  const int kg   = lane >> 4;

  // ---------------- stage 2: conv1 as implicit GEMM M=64 K=576 N=128 -------
  {
    v8f acc[4] = {};
    const _Float16* bbase = w1T + (size_t)(wv * 16 + lr) * 576 + kg * 8;
    for (int ks = 0; ks < 18; ++ks) {
      Frag bf;
      const _Float16* bp = bbase + ks * 32;
      bf.q[0] = *(const f4*)bp;
      bf.q[1] = *(const f4*)(bp + 16);
      #pragma unroll
      for (int mt = 0; mt < 4; ++mt) {
        Frag af;
        const int m = mt * 16 + lr;
        const int oy = m >> 3, ox = m & 7;
        #pragma unroll
        for (int r = 0; r < 2; ++r) {
          const int K = ks * 32 + r * 16 + kg * 8;   // 8-aligned: run stays in 1 pixel
          const int kidx = K >> 6;                    // ky*3+kx
          const int cin  = K & 63;
          const int ky = kidx / 3, kx = kidx - ky * 3;
          const int iy = oy * 2 + ky, ix = ox * 2 + kx;   // SAME pad: high edge only
          if (iy < 16 && ix < 16) {
            af.q[r] = *(const f4*)&smx[(iy * 16 + ix) * 64 + cin];
          } else {
            f4 z = {0.f, 0.f, 0.f, 0.f};
            af.q[r] = z;
          }
        }
        acc[mt] = __builtin_amdgcn_wmma_f32_16x16x32_f16(
            false, af.v, false, bf.v, (short)0, acc[mt], false, false);
      }
    }
    const float bias = b1[wv * 16 + lr];
    #pragma unroll
    for (int mt = 0; mt < 4; ++mt)
      #pragma unroll
      for (int vv = 0; vv < 8; ++vv) {
        const int m = mt * 16 + vv + kg * 8;
        smy[m * 128 + wv * 16 + lr] = (_Float16)fmaxf(acc[mt][vv] + bias, 0.0f);
      }
  }
  __syncthreads();

  // ---------------- stage 3: conv2 as implicit GEMM M=16 K=1152 N=128 ------
  {
    v8f acc = {};
    const _Float16* bbase = w2T + (size_t)(wv * 16 + lr) * 1152 + kg * 8;
    const int oy = lr >> 2, ox = lr & 3;
    for (int ks = 0; ks < 36; ++ks) {
      Frag bf;
      const _Float16* bp = bbase + ks * 32;
      bf.q[0] = *(const f4*)bp;
      bf.q[1] = *(const f4*)(bp + 16);
      Frag af;
      #pragma unroll
      for (int r = 0; r < 2; ++r) {
        const int K = ks * 32 + r * 16 + kg * 8;
        const int kidx = K >> 7;
        const int cin  = K & 127;
        const int ky = kidx / 3, kx = kidx - ky * 3;
        const int iy = oy * 2 + ky, ix = ox * 2 + kx;
        if (iy < 8 && ix < 8) {
          af.q[r] = *(const f4*)&smy[(iy * 8 + ix) * 128 + cin];
        } else {
          f4 z = {0.f, 0.f, 0.f, 0.f};
          af.q[r] = z;
        }
      }
      acc = __builtin_amdgcn_wmma_f32_16x16x32_f16(
          false, af.v, false, bf.v, (short)0, acc, false, false);
    }
    const float bias = b2[wv * 16 + lr];
    _Float16* out = y2out + (size_t)box * 2048;
    #pragma unroll
    for (int vv = 0; vv < 8; ++vv) {
      const int m = vv + kg * 8;   // flatten order (oy*4+ox)*128+c matches reshape
      out[m * 128 + wv * 16 + lr] = (_Float16)fmaxf(acc[vv] + bias, 0.0f);
    }
  }
}

// ---------------------------------------------------------------------------
// Generic WMMA GEMM: C[M][N] = relu(A[M][K] @ B + bias), B given as BT [N][K].
// Block = 8 waves = 64x128 output tile. grid = (N/128, M/64).
// ---------------------------------------------------------------------------
__global__ void __launch_bounds__(256)
gemm_bias_relu(const _Float16* __restrict__ A,
               const _Float16* __restrict__ BT,
               const float* __restrict__ bias,
               _Float16* __restrict__ C,
               int M, int N, int K)
{
  const int t = threadIdx.x;
  const int lane = t & 31, wv = t >> 5;
  const int lr = lane & 15, kg = lane >> 4;
  const int Ncol = blockIdx.x * 128 + wv * 16;
  const int Moff = blockIdx.y * 64;

  v8f acc[4] = {};
  const _Float16* bbase = BT + (size_t)(Ncol + lr) * K + kg * 8;
  const int ksteps = K >> 5;
  for (int ks = 0; ks < ksteps; ++ks) {
    Frag bf;
    const _Float16* bp = bbase + ks * 32;
    bf.q[0] = *(const f4*)bp;
    bf.q[1] = *(const f4*)(bp + 16);
    #pragma unroll
    for (int mt = 0; mt < 4; ++mt) {
      Frag af;
      const _Float16* ap = A + (size_t)(Moff + mt * 16 + lr) * K + ks * 32 + kg * 8;
      af.q[0] = *(const f4*)ap;          // K run [kg*8, +8)
      af.q[1] = *(const f4*)(ap + 16);   // K run [16+kg*8, +8)
      acc[mt] = __builtin_amdgcn_wmma_f32_16x16x32_f16(
          false, af.v, false, bf.v, (short)0, acc[mt], false, false);
    }
  }
  const float bs = bias[Ncol + lr];
  #pragma unroll
  for (int mt = 0; mt < 4; ++mt)
    #pragma unroll
    for (int vv = 0; vv < 8; ++vv) {
      const int row = Moff + mt * 16 + vv + kg * 8;
      C[(size_t)row * N + Ncol + lr] = (_Float16)fmaxf(acc[mt][vv] + bs, 0.0f);
    }
}

// ---------------------------------------------------------------------------
// Head: per box, score (1) + regressions (4) via 512-length dots, box decode.
// Output layout: [scores 4096][regressions 4096*4][bboxes 4096*4]
// ---------------------------------------------------------------------------
__global__ void __launch_bounds__(256)
head_kernel(const _Float16* __restrict__ x,
            const float* __restrict__ bboxes,
            const int* __restrict__ cls,
            const float* __restrict__ reg_w, const float* __restrict__ reg_b,
            const float* __restrict__ score_w, const float* __restrict__ score_b,
            float* __restrict__ out)
{
  const int i = blockIdx.x * blockDim.x + threadIdx.x;
  if (i >= NBOX) return;
  const _Float16* xr = x + (size_t)i * 512;
  const int c = cls[i];
  float r0 = reg_b[c * 4 + 0], r1 = reg_b[c * 4 + 1];
  float r2 = reg_b[c * 4 + 2], r3 = reg_b[c * 4 + 3];
  float sc = score_b[c];
  for (int k = 0; k < 512; ++k) {
    const float xv = (float)xr[k];
    sc += xv * score_w[k * 3 + c];
    const float* rw = reg_w + (size_t)k * 12 + c * 4;
    r0 += xv * rw[0];
    r1 += xv * rw[1];
    r2 += xv * rw[2];
    r3 += xv * rw[3];
  }
  const float y1 = bboxes[i * 4 + 0], x1 = bboxes[i * 4 + 1];
  const float y2 = bboxes[i * 4 + 2], x2 = bboxes[i * 4 + 3];
  const float h = y2 - y1, w = x2 - x1;
  const float cy = y1 + 0.5f * h + r0 * h;
  const float cx = x1 + 0.5f * w + r1 * w;
  const float nh = h * expf(r2);
  const float nw = w * expf(r3);

  out[i] = sc;
  float* reg_out = out + NBOX;
  reg_out[i * 4 + 0] = r0;
  reg_out[i * 4 + 1] = r1;
  reg_out[i * 4 + 2] = r2;
  reg_out[i * 4 + 3] = r3;
  float* box_out = out + NBOX + NBOX * 4;
  box_out[i * 4 + 0] = cy - 0.5f * nh;
  box_out[i * 4 + 1] = cx - 0.5f * nw;
  box_out[i * 4 + 2] = cy + 0.5f * nh;
  box_out[i * 4 + 3] = cx + 0.5f * nw;
}

// ---------------------------------------------------------------------------
extern "C" void kernel_launch(void* const* d_in, const int* in_sizes, int n_in,
                              void* d_out, int out_size, void* d_ws, size_t ws_size,
                              hipStream_t stream) {
  const float* features = (const float*)d_in[0];
  const float* proposal = (const float*)d_in[1];
  const float* bboxes   = (const float*)d_in[2];
  const int*   box_idx  = (const int*)d_in[3];
  const int*   cls      = (const int*)d_in[4];
  const float* conv1_w  = (const float*)d_in[5];
  const float* conv1_b  = (const float*)d_in[6];
  const float* conv2_w  = (const float*)d_in[7];
  const float* conv2_b  = (const float*)d_in[8];
  const float* d1_w     = (const float*)d_in[9];
  const float* d1_b     = (const float*)d_in[10];
  const float* d2_w     = (const float*)d_in[11];
  const float* d2_b     = (const float*)d_in[12];
  const float* reg_w    = (const float*)d_in[13];
  const float* reg_b    = (const float*)d_in[14];
  const float* score_w  = (const float*)d_in[15];
  const float* score_b  = (const float*)d_in[16];
  float* out = (float*)d_out;

  char* ws = (char*)d_ws;
  size_t off = 0;
  auto alloc = [&](size_t bytes) -> char* {
    char* p = ws + off;
    off += (bytes + 255) & ~(size_t)255;
    return p;
  };
  _Float16* w1T = (_Float16*)alloc((size_t)128 * 576 * 2);
  _Float16* w2T = (_Float16*)alloc((size_t)128 * 1152 * 2);
  _Float16* d1T = (_Float16*)alloc((size_t)512 * 2048 * 2);
  _Float16* d2T = (_Float16*)alloc((size_t)512 * 512 * 2);
  _Float16* y2  = (_Float16*)alloc((size_t)NBOX * 2048 * 2);
  _Float16* a1  = (_Float16*)alloc((size_t)NBOX * 512 * 2);
  _Float16* a2  = (_Float16*)alloc((size_t)NBOX * 512 * 2);

  convert_transpose<<<(576 * 128 + 255) / 256, 256, 0, stream>>>(conv1_w, w1T, 576, 128);
  convert_transpose<<<(1152 * 128 + 255) / 256, 256, 0, stream>>>(conv2_w, w2T, 1152, 128);
  convert_transpose<<<(2048 * 512 + 255) / 256, 256, 0, stream>>>(d1_w, d1T, 2048, 512);
  convert_transpose<<<(512 * 512 + 255) / 256, 256, 0, stream>>>(d2_w, d2T, 512, 512);

  fused_crop_conv<<<NBOX, 256, 0, stream>>>(features, proposal, bboxes, box_idx,
                                            w1T, conv1_b, w2T, conv2_b, y2);
  gemm_bias_relu<<<dim3(512 / 128, 4096 / 64), 256, 0, stream>>>(y2, d1T, d1_b, a1, 4096, 512, 2048);
  gemm_bias_relu<<<dim3(512 / 128, 4096 / 64), 256, 0, stream>>>(a1, d2T, d2_b, a2, 4096, 512, 512);
  head_kernel<<<NBOX / 256, 256, 0, stream>>>(a2, bboxes, cls, reg_w, reg_b, score_w, score_b, out);
}
